// Decoder_block_4827543241240
// MI455X (gfx1250) — matrix operand order repaired and checked by reading, not compiled, not run on hardware
//
#include <hip/hip_runtime.h>
#include <hip/hip_bf16.h>

// MI455X (gfx1250) implementation.
// All matmuls (projections, FFN, QK^T, PV) run on v_wmma_f32_16x16x32_f16
// with fp32 accumulation. Attention is flash-style with online softmax and
// causal pyramid-block skipping. Workspace use ~= 400 MB.

typedef __attribute__((ext_vector_type(16))) _Float16 v16h;
typedef __attribute__((ext_vector_type(8)))  _Float16 half8;
typedef __attribute__((ext_vector_type(8)))  float    v8f;

#define BATCH 16
#define SEQ   1024
#define DM    512
#define NHEAD 8
#define DH    64
#define KTOT  1792   // 1024 + 512 + 256
#define NTOK  (BATCH * SEQ)          // 16384
#define BH    (BATCH * NHEAD)        // 128

__device__ __forceinline__ v8f wmma16(v16h a, v16h b, v8f c) {
  return __builtin_amdgcn_wmma_f32_16x16x32_f16(
      /*neg_a=*/false, a, /*neg_b=*/false, b,
      /*c_mod=*/(short)0, c, /*reuse_a=*/false, /*reuse_b=*/false);
}

// A fragment (16x32 f16, row-major source, tile base pre-offset).
// lane m = lane&15 holds row m; element e -> k = (e/8)*16 + (lane>>4)*8 + e%8.
__device__ __forceinline__ v16h load_a_frag(const _Float16* A, int lda) {
  const int lane = threadIdx.x & 31;
  const int m = lane & 15;
  const int hi = lane >> 4;
  const _Float16* p = A + (size_t)m * lda + hi * 8;
  half8 lo = *reinterpret_cast<const half8*>(p);
  half8 hh = *reinterpret_cast<const half8*>(p + 16);
  v16h r;
#pragma unroll
  for (int i = 0; i < 8; ++i) { r[i] = lo[i]; r[i + 8] = hh[i]; }
  return r;
}

// B fragment (32x16 f16) from a TRANSPOSED source Bt (N x K row-major):
// lane n = lane&15 holds column n; element e -> k = (e/8)*16 + (lane>>4)*8 + e%8.
__device__ __forceinline__ v16h load_bt_frag(const _Float16* Bt, int ldb) {
  const int lane = threadIdx.x & 31;
  const int n = lane & 15;
  const int hi = lane >> 4;
  const _Float16* p = Bt + (size_t)n * ldb + hi * 8;
  half8 a = *reinterpret_cast<const half8*>(p);
  half8 b = *reinterpret_cast<const half8*>(p + 16);
  v16h r;
#pragma unroll
  for (int i = 0; i < 8; ++i) { r[i] = a[i]; r[i + 8] = b[i]; }
  return r;
}

// ---------------- weight transpose + f16 convert -------------------------
__global__ void wconv_kernel(const float* __restrict__ W, _Float16* __restrict__ Wt) {
  int idx = blockIdx.x * 256 + threadIdx.x;       // 512*512
  int n = idx >> 9, k = idx & 511;
  Wt[idx] = (_Float16)W[(size_t)k * DM + n];      // Wt[n][k] = W[k][n]
}

// ---------------- fused embedding gather + LayerNorm ---------------------
__global__ __launch_bounds__(256) void embed_ln_kernel(
    const int* __restrict__ in_ex, const int* __restrict__ in_cat,
    const int* __restrict__ in_res, const float* __restrict__ in_pos,
    const float* __restrict__ E_res, const float* __restrict__ E_ex,
    const float* __restrict__ E_cat,
    const float* __restrict__ g, const float* __restrict__ be,
    float* __restrict__ y, _Float16* __restrict__ yh) {
  const int w = threadIdx.x >> 5, lane = threadIdx.x & 31;
  const int tok = blockIdx.x * 8 + w;             // B*S tokens
  const int res = in_res[tok], ex = in_ex[tok], cat = in_cat[tok];
  const float* er = E_res + (size_t)res * DM;
  const float* ee = E_ex + (size_t)(ex + 10000 * res) * DM;
  const float* ec = E_cat + (size_t)(cat + 300 * res) * DM;
  const float* pp = in_pos + (size_t)tok * DM;
  float v[16]; float s = 0.f;
#pragma unroll
  for (int j = 0; j < 16; ++j) {
    int d = lane + j * 32;
    v[j] = er[d] + ee[d] + ec[d] + pp[d];
    s += v[j];
  }
#pragma unroll
  for (int o = 16; o; o >>= 1) s += __shfl_xor(s, o, 32);
  float mean = s * (1.f / DM);
  float q = 0.f;
#pragma unroll
  for (int j = 0; j < 16; ++j) { float d = v[j] - mean; q += d * d; }
#pragma unroll
  for (int o = 16; o; o >>= 1) q += __shfl_xor(q, o, 32);
  float rstd = rsqrtf(q * (1.f / DM) + 1e-5f);
  float* yo = y + (size_t)tok * DM;
  _Float16* yho = yh + (size_t)tok * DM;
#pragma unroll
  for (int j = 0; j < 16; ++j) {
    int d = lane + j * 32;
    float o2 = (v[j] - mean) * rstd * g[d] + be[d];
    yo[d] = o2;
    yho[d] = (_Float16)o2;
  }
}

// ---------------- generic LayerNorm (optional f32/f16 outputs) -----------
__global__ __launch_bounds__(256) void ln_kernel(
    const float* __restrict__ x, const float* __restrict__ g,
    const float* __restrict__ be, float* __restrict__ y,
    _Float16* __restrict__ yh) {
  const int w = threadIdx.x >> 5, lane = threadIdx.x & 31;
  const int tok = blockIdx.x * 8 + w;
  const float* xp = x + (size_t)tok * DM;
  float v[16]; float s = 0.f;
#pragma unroll
  for (int j = 0; j < 16; ++j) { v[j] = xp[lane + j * 32]; s += v[j]; }
#pragma unroll
  for (int o = 16; o; o >>= 1) s += __shfl_xor(s, o, 32);
  float mean = s * (1.f / DM);
  float q = 0.f;
#pragma unroll
  for (int j = 0; j < 16; ++j) { float d = v[j] - mean; q += d * d; }
#pragma unroll
  for (int o = 16; o; o >>= 1) q += __shfl_xor(q, o, 32);
  float rstd = rsqrtf(q * (1.f / DM) + 1e-5f);
#pragma unroll
  for (int j = 0; j < 16; ++j) {
    int d = lane + j * 32;
    float o2 = (v[j] - mean) * rstd * g[d] + be[d];
    if (y)  y[(size_t)tok * DM + d] = o2;
    if (yh) yh[(size_t)tok * DM + d] = (_Float16)o2;
  }
}

// ---------------- tiled WMMA GEMM: C = A @ Bt^T + bias (+resid)(relu) ----
// A: M x K f16 row-major. Bt: N x K f16 row-major (= W transposed).
// WG = 256 threads = 8 waves, block tile 128x128, wave tile 32x64.
__global__ __launch_bounds__(256) void gemm_wmma_kernel(
    const _Float16* __restrict__ A, const _Float16* __restrict__ Bt,
    const float* __restrict__ bias, const float* __restrict__ resid,
    float* __restrict__ C, _Float16* __restrict__ Ch,
    int M, int N, int K, int relu) {
  const int w = threadIdx.x >> 5;
  const int wm = w & 3, wn = w >> 2;
  const int m0 = blockIdx.y * 128 + wm * 32;
  const int n0 = blockIdx.x * 128 + wn * 64;
  const int lane = threadIdx.x & 31;
  const int nn = lane & 15, hi = lane >> 4;
  v8f acc[2][4] = {};
  for (int kk = 0; kk < K; kk += 32) {
    v16h a0 = load_a_frag(A + (size_t)m0 * K + kk, K);
    v16h a1 = load_a_frag(A + (size_t)(m0 + 16) * K + kk, K);
#pragma unroll
    for (int j = 0; j < 4; ++j) {
      v16h b = load_bt_frag(Bt + (size_t)(n0 + j * 16) * K + kk, K);
      acc[0][j] = wmma16(a0, b, acc[0][j]);
      acc[1][j] = wmma16(a1, b, acc[1][j]);
    }
  }
#pragma unroll
  for (int i = 0; i < 2; ++i)
#pragma unroll
    for (int j = 0; j < 4; ++j)
#pragma unroll
      for (int r = 0; r < 8; ++r) {
        int m = m0 + i * 16 + r + hi * 8;
        int n = n0 + j * 16 + nn;
        float v = acc[i][j][r] + bias[n];
        if (resid) v += resid[(size_t)m * N + n];
        if (relu) v = fmaxf(v, 0.f);
        C[(size_t)m * N + n] = v;
        if (Ch) Ch[(size_t)m * N + n] = (_Float16)v;
      }
}

// ---------------- head split (Q, pre-scaled by 1/sqrt(dh)) ---------------
__global__ void split_q_kernel(const float* __restrict__ q, _Float16* __restrict__ Qh) {
  size_t idx = (size_t)blockIdx.x * 256 + threadIdx.x;   // BH*S*64
  int d = idx & 63;
  size_t t = idx >> 6;            // bh*S + s
  int s = t & (SEQ - 1);
  size_t bh = t >> 10;
  int b = (int)(bh >> 3), h = (int)(bh & 7);
  Qh[idx] = (_Float16)(q[((size_t)(b * SEQ + s)) * DM + h * DH + d] * 0.125f);
}

// ---------------- pyramid pooling of K (row-major) and V (transposed) ----
__device__ __forceinline__ float pool_val(const float* __restrict__ kv,
                                          int b, int h, int key, int d) {
  int t0, f;
  if (key < 1024)      { t0 = key;                f = 1; }
  else if (key < 1536) { t0 = (key - 1024) * 2;   f = 2; }
  else                 { t0 = (key - 1536) * 4;   f = 4; }
  float s = 0.f;
  for (int u = 0; u < f; ++u)
    s += kv[((size_t)(b * SEQ + t0 + u)) * DM + h * DH + d];
  return s / (float)f;
}

__global__ void pool_k_kernel(const float* __restrict__ kv, _Float16* __restrict__ Kp) {
  size_t idx = (size_t)blockIdx.x * 256 + threadIdx.x;   // BH*KTOT*64
  int d = idx & 63;
  size_t t = idx >> 6;
  int key = (int)(t % KTOT);
  size_t bh = t / KTOT;
  int b = (int)(bh >> 3), h = (int)(bh & 7);
  Kp[idx] = (_Float16)pool_val(kv, b, h, key, d);        // (BH, KTOT, 64)
}

__global__ void pool_v_kernel(const float* __restrict__ kv, _Float16* __restrict__ Vt) {
  size_t idx = (size_t)blockIdx.x * 256 + threadIdx.x;   // BH*KTOT*64
  int d = idx & 63;
  size_t t = idx >> 6;
  int key = (int)(t % KTOT);
  size_t bh = t / KTOT;
  int b = (int)(bh >> 3), h = (int)(bh & 7);
  Vt[bh * (size_t)DH * KTOT + (size_t)d * KTOT + key] =  // (BH, 64, KTOT)
      (_Float16)pool_val(kv, b, h, key, d);
}

// ---------------- flash attention over pyramid keys ----------------------
// grid = (S/64, BH), block = 128 (4 waves), one wave per 16-query tile.
__global__ __launch_bounds__(128) void attn_kernel(
    const _Float16* __restrict__ Qh, const _Float16* __restrict__ Kp,
    const _Float16* __restrict__ Vt, float* __restrict__ Oh) {
  __shared__ _Float16 plds_all[4 * 16 * 32];
  const int w = threadIdx.x >> 5, lane = threadIdx.x & 31;
  const int nn = lane & 15, hi = lane >> 4;
  const int bh = blockIdx.y;
  const int q0 = blockIdx.x * 64 + w * 16;
  const _Float16* Qb = Qh + ((size_t)bh * SEQ + q0) * DH;
  const v16h qa0 = load_a_frag(Qb, DH);
  const v16h qa1 = load_a_frag(Qb + 32, DH);
  const _Float16* Kbase = Kp + (size_t)bh * KTOT * DH;
  const _Float16* Vbase = Vt + (size_t)bh * DH * KTOT;
  _Float16* pl = plds_all + w * 16 * 32;

  v8f o[4] = {};
  float mrun[8], lrun[8];
#pragma unroll
  for (int r = 0; r < 8; ++r) { mrun[r] = -3.0e38f; lrun[r] = 0.f; }

  const int imax = q0 + 15;
  const int lim0 = imax + 1;                        // level0: i >= key
  const int lim1 = 1024 + ((imax - 1) >> 1) + 1;    // level1: i >= 2j+1
  const int lim2 = 1536 + ((imax - 3) >> 2) + 1;    // level2: i >= 4j+3

  for (int key0 = 0; key0 < KTOT; key0 += 32) {
    if (key0 < 1024)      { if (key0 >= lim0) { key0 = 1024 - 32; continue; } }
    else if (key0 < 1536) { if (key0 >= lim1) { key0 = 1536 - 32; continue; } }
    else                  { if (key0 >= lim2) break; }

    // scores: two 16x16 tiles (keys key0..+15, key0+16..+31)
    const _Float16* Kb = Kbase + (size_t)key0 * DH;
    v8f s0 = {}, s1 = {};
    {
      v16h b00 = load_bt_frag(Kb, DH);
      v16h b01 = load_bt_frag(Kb + 32, DH);
      s0 = wmma16(qa0, b00, s0);
      s0 = wmma16(qa1, b01, s0);
      v16h b10 = load_bt_frag(Kb + 16 * DH, DH);
      v16h b11 = load_bt_frag(Kb + 16 * DH + 32, DH);
      s1 = wmma16(qa0, b10, s1);
      s1 = wmma16(qa1, b11, s1);
    }

    const int k0i = key0 + nn, k1i = key0 + 16 + nn;
    const int thr0 = (k0i < 1024) ? k0i : (k0i < 1536 ? 2 * (k0i - 1024) + 1
                                                      : 4 * (k0i - 1536) + 3);
    const int thr1 = (k1i < 1024) ? k1i : (k1i < 1536 ? 2 * (k1i - 1024) + 1
                                                      : 4 * (k1i - 1536) + 3);
    float alpha[8];
#pragma unroll
    for (int r = 0; r < 8; ++r) {
      const int i = q0 + r + hi * 8;
      float sv0 = (i >= thr0) ? s0[r] : -3.0e38f;
      float sv1 = (i >= thr1) ? s1[r] : -3.0e38f;
      float bm = fmaxf(sv0, sv1);
#pragma unroll
      for (int off = 1; off < 16; off <<= 1) bm = fmaxf(bm, __shfl_xor(bm, off, 32));
      const float mnew = fmaxf(mrun[r], bm);
      const float p0 = __expf(sv0 - mnew);
      const float p1 = __expf(sv1 - mnew);
      float ps = p0 + p1;
#pragma unroll
      for (int off = 1; off < 16; off <<= 1) ps += __shfl_xor(ps, off, 32);
      const float a = __expf(mrun[r] - mnew);
      lrun[r] = lrun[r] * a + ps;
      mrun[r] = mnew;
      alpha[r] = a;
      pl[(r + hi * 8) * 32 + nn]      = (_Float16)p0;
      pl[(r + hi * 8) * 32 + 16 + nn] = (_Float16)p1;
    }
    asm volatile("s_wait_dscnt 0" ::: "memory");   // in-wave LDS RAW fence
    const v16h pa = load_a_frag(pl, 32);
#pragma unroll
    for (int t = 0; t < 4; ++t) {
#pragma unroll
      for (int r = 0; r < 8; ++r) o[t][r] *= alpha[r];
      v16h vb = load_bt_frag(Vbase + (size_t)(t * 16) * KTOT + key0, KTOT);
      o[t] = wmma16(pa, vb, o[t]);
    }
  }

  float* Ob = Oh + ((size_t)bh * SEQ + q0) * DH;
#pragma unroll
  for (int t = 0; t < 4; ++t)
#pragma unroll
    for (int r = 0; r < 8; ++r)
      Ob[(size_t)(r + hi * 8) * DH + t * 16 + nn] = o[t][r] / lrun[r];
}

// ---------------- merge heads: (BH,S,64) f32 -> (B,S,D) f16 --------------
__global__ void merge_h_kernel(const float* __restrict__ Oh, _Float16* __restrict__ Ah) {
  size_t idx = (size_t)blockIdx.x * 256 + threadIdx.x;   // B*S*D
  int dg = idx & 511;
  size_t t = idx >> 9;            // b*S + s
  int h = dg >> 6, d = dg & 63;
  size_t b = t >> 10; int s = (int)(t & (SEQ - 1));
  Ah[idx] = (_Float16)Oh[((b * NHEAD + h) * (size_t)SEQ + s) * DH + d];
}

// -------------------------------------------------------------------------
extern "C" void kernel_launch(void* const* d_in, const int* in_sizes, int n_in,
                              void* d_out, int out_size, void* d_ws, size_t ws_size,
                              hipStream_t stream) {
  (void)in_sizes; (void)n_in; (void)out_size; (void)ws_size;
  const int*   in_ex  = (const int*)d_in[0];
  const int*   in_cat = (const int*)d_in[1];
  const int*   in_res = (const int*)d_in[2];
  const float* in_pos = (const float*)d_in[3];
  const float* en_out = (const float*)d_in[4];
  const float* E_res  = (const float*)d_in[5];
  const float* E_ex   = (const float*)d_in[6];
  const float* E_cat  = (const float*)d_in[7];
  const float* g1 = (const float*)d_in[8],  *be1 = (const float*)d_in[9];
  const float* g2 = (const float*)d_in[10], *be2 = (const float*)d_in[11];
  const float* g3 = (const float*)d_in[12], *be3 = (const float*)d_in[13];
  // weight order: Wq1 Wk1 Wv1 Wo1 Wq2 Wk2 Wv2 Wo2 fW1 fW2
  const float* W[10]  = { (const float*)d_in[14], (const float*)d_in[16],
                          (const float*)d_in[18], (const float*)d_in[20],
                          (const float*)d_in[22], (const float*)d_in[24],
                          (const float*)d_in[26], (const float*)d_in[28],
                          (const float*)d_in[30], (const float*)d_in[32] };
  const float* Bv[10] = { (const float*)d_in[15], (const float*)d_in[17],
                          (const float*)d_in[19], (const float*)d_in[21],
                          (const float*)d_in[23], (const float*)d_in[25],
                          (const float*)d_in[27], (const float*)d_in[29],
                          (const float*)d_in[31], (const float*)d_in[33] };

  // ---- workspace carve-out ----
  char* p = (char*)d_ws;
  auto alloc = [&](size_t bytes) -> void* {
    void* r = (void*)p;
    p += (bytes + 255) & ~(size_t)255;
    return r;
  };
  const size_t BSD = (size_t)NTOK * DM;          // 8,388,608
  _Float16* Wt[10];
  for (int i = 0; i < 10; ++i) Wt[i] = (_Float16*)alloc((size_t)DM * DM * 2);
  float*    x1  = (float*)alloc(BSD * 4);
  _Float16* x1h = (_Float16*)alloc(BSD * 2);
  float*    x2  = (float*)alloc(BSD * 4);
  _Float16* x2h = (_Float16*)alloc(BSD * 2);
  float*    x3  = (float*)alloc(BSD * 4);
  float*    x4  = (float*)alloc(BSD * 4);
  _Float16* x4h = (_Float16*)alloc(BSD * 2);
  _Float16* enh = (_Float16*)alloc(BSD * 2);
  float*    t0  = (float*)alloc(BSD * 4);
  _Float16* Qh  = (_Float16*)alloc((size_t)BH * SEQ * DH * 2);
  _Float16* Kp  = (_Float16*)alloc((size_t)BH * KTOT * DH * 2);
  _Float16* Vt  = (_Float16*)alloc((size_t)BH * KTOT * DH * 2);
  float*    OhF = (float*)alloc((size_t)BH * SEQ * DH * 4);
  _Float16* Ah  = (_Float16*)alloc(BSD * 2);
  _Float16* hh  = (_Float16*)alloc(BSD * 2);

  const dim3 ggrid(DM / 128, NTOK / 128);        // (4, 128)
  const dim3 agrid(SEQ / 64, BH);                // (16, 128)
  const int  lnBlocks   = NTOK / 8;              // 2048
  const int  qBlocks    = (int)(((size_t)BH * SEQ * DH) / 256);
  const int  poolBlocks = (int)(((size_t)BH * KTOT * DH) / 256);
  const int  bsdBlocks  = (int)(BSD / 256);

  // 0) weights -> transposed f16
  for (int i = 0; i < 10; ++i)
    wconv_kernel<<<DM * DM / 256, 256, 0, stream>>>(W[i], Wt[i]);

  // 1) embed + LN1 -> x1 (f32 + f16)
  embed_ln_kernel<<<lnBlocks, 256, 0, stream>>>(in_ex, in_cat, in_res, in_pos,
                                                E_res, E_ex, E_cat, g1, be1, x1, x1h);

  // 2) self attention on x1
  gemm_wmma_kernel<<<ggrid, 256, 0, stream>>>(x1h, Wt[0], Bv[0], nullptr, t0, nullptr, NTOK, DM, DM, 0);
  split_q_kernel<<<qBlocks, 256, 0, stream>>>(t0, Qh);
  gemm_wmma_kernel<<<ggrid, 256, 0, stream>>>(x1h, Wt[1], Bv[1], nullptr, t0, nullptr, NTOK, DM, DM, 0);
  pool_k_kernel<<<poolBlocks, 256, 0, stream>>>(t0, Kp);
  gemm_wmma_kernel<<<ggrid, 256, 0, stream>>>(x1h, Wt[2], Bv[2], nullptr, t0, nullptr, NTOK, DM, DM, 0);
  pool_v_kernel<<<poolBlocks, 256, 0, stream>>>(t0, Vt);
  attn_kernel<<<agrid, 128, 0, stream>>>(Qh, Kp, Vt, OhF);
  merge_h_kernel<<<bsdBlocks, 256, 0, stream>>>(OhF, Ah);
  gemm_wmma_kernel<<<ggrid, 256, 0, stream>>>(Ah, Wt[3], Bv[3], x1, x2, x2h, NTOK, DM, DM, 0);

  // 3) cross attention: q from x2, kv from LN(en_out)
  ln_kernel<<<lnBlocks, 256, 0, stream>>>(en_out, g2, be2, nullptr, enh);
  gemm_wmma_kernel<<<ggrid, 256, 0, stream>>>(x2h, Wt[4], Bv[4], nullptr, t0, nullptr, NTOK, DM, DM, 0);
  split_q_kernel<<<qBlocks, 256, 0, stream>>>(t0, Qh);
  gemm_wmma_kernel<<<ggrid, 256, 0, stream>>>(enh, Wt[5], Bv[5], nullptr, t0, nullptr, NTOK, DM, DM, 0);
  pool_k_kernel<<<poolBlocks, 256, 0, stream>>>(t0, Kp);
  gemm_wmma_kernel<<<ggrid, 256, 0, stream>>>(enh, Wt[6], Bv[6], nullptr, t0, nullptr, NTOK, DM, DM, 0);
  pool_v_kernel<<<poolBlocks, 256, 0, stream>>>(t0, Vt);
  attn_kernel<<<agrid, 128, 0, stream>>>(Qh, Kp, Vt, OhF);
  merge_h_kernel<<<bsdBlocks, 256, 0, stream>>>(OhF, Ah);
  gemm_wmma_kernel<<<ggrid, 256, 0, stream>>>(Ah, Wt[7], Bv[7], x2, x3, nullptr, NTOK, DM, DM, 0);

  // 4) LN3 + FFN with residual from x4
  ln_kernel<<<lnBlocks, 256, 0, stream>>>(x3, g3, be3, x4, x4h);
  gemm_wmma_kernel<<<ggrid, 256, 0, stream>>>(x4h, Wt[8], Bv[8], nullptr, t0, hh, NTOK, DM, DM, 1);
  gemm_wmma_kernel<<<ggrid, 256, 0, stream>>>(hh, Wt[9], Bv[9], x4, (float*)d_out, nullptr, NTOK, DM, DM, 0);
}
